// GCN_3530463117755
// MI455X (gfx1250) — compile-verified, matrix operand order checked by reading, not run on hardware
//
#include <hip/hip_runtime.h>
#include <hip/hip_bf16.h>

// ---------------------------------------------------------------------------
// GCN forward for gfx1250 (MI455X):
//  - dense transforms via v_wmma_f32_16x16x32_f16 (f16 A/B, fp32 acc)
//  - message passing via L2-resident gather + global_atomic_add_f32 scatter
// ---------------------------------------------------------------------------

#define NNODES 100000
#define NEDGES 1600000
#define NGRAPH 64
#define DHID   128
#define DOUT   10

typedef __attribute__((ext_vector_type(16))) _Float16 v16h;
typedef __attribute__((ext_vector_type(8)))  _Float16 v8h;
typedef __attribute__((ext_vector_type(8)))  float    v8f;

// -------------------------- small utility kernels --------------------------

__global__ void k_fill(float* __restrict__ p, float v, int n) {
    int i = blockIdx.x * blockDim.x + threadIdx.x;
    if (i < n) p[i] = v;
}

// deg[dst[e]] += 1  (self-loop contribution pre-seeded as deg=1.0)
__global__ void k_degree(const int* __restrict__ dst, float* __restrict__ deg, int e_cnt) {
    int i = blockIdx.x * blockDim.x + threadIdx.x;
    if (i < e_cnt) atomicAdd(&deg[dst[i]], 1.0f);
}

__global__ void k_rsqrt_inplace(float* __restrict__ p, int n) {
    int i = blockIdx.x * blockDim.x + threadIdx.x;
    if (i < n) p[i] = rsqrtf(fmaxf(p[i], 1.0f));   // deg >= 1 always (self loops)
}

// out[n][c] = b[c]   (scatter target starts at bias => segment_sum + b)
__global__ void k_bias_init(float* __restrict__ out, const float* __restrict__ b, int n) {
    int i = blockIdx.x * blockDim.x + threadIdx.x;
    if (i < n) out[i] = b[i & (DHID - 1)];
}

__global__ void k_relu(float* __restrict__ p, int n) {
    int i = blockIdx.x * blockDim.x + threadIdx.x;
    if (i < n) p[i] = fmaxf(p[i], 0.0f);
}

// ------------------------------ WMMA GEMM ----------------------------------
// Mout[N x 128] = H[N x 128] @ W[128 x 128]
// One wave computes a 16-row x 128-col strip. W is staged per block into LDS,
// transposed and converted to f16, so B fragments are contiguous ds_load_b128.

__global__ void __launch_bounds__(256)
k_gemm_wmma(const float* __restrict__ H, const float* __restrict__ W,
            float* __restrict__ Mout, int nrows) {
    __shared__ _Float16 WT[DHID * DHID];   // WT[n][k] = (f16) W[k][n]  (32 KB)

    for (int idx = threadIdx.x; idx < DHID * DHID; idx += blockDim.x) {
        int k = idx >> 7, n = idx & 127;
        WT[n * DHID + k] = (_Float16)W[idx];
    }
    __syncthreads();

    const int lane = threadIdx.x & 31;
    const int wave = threadIdx.x >> 5;
    const int tile = blockIdx.x * 8 + wave;
    const int ntiles = (nrows + 15) >> 4;
    if (tile >= ntiles) return;

    const int lrow = lane & 15;       // row within 16-row tile (A) / column (B,C)
    const int hi   = lane >> 4;       // upper/lower half-wave K split

    int arow = tile * 16 + lrow;
    if (arow >= nrows) arow = nrows - 1;   // clamp (harmless duplicate read)

    v8f acc[8];
#pragma unroll
    for (int i = 0; i < 8; ++i) acc[i] = (v8f)(0.0f);

#pragma unroll
    for (int kt = 0; kt < 4; ++kt) {       // K = 128 in 4 x 32 steps
        // ---- A fragment: f32 global -> f16 regs, layout per ISA 16-bit A 16x32
        v16h a;
        const float* hp = H + (size_t)arow * DHID + kt * 32 + hi * 8;
#pragma unroll
        for (int j = 0; j < 8; ++j) a[j]     = (_Float16)hp[j];        // K: base..base+7
#pragma unroll
        for (int j = 0; j < 8; ++j) a[8 + j] = (_Float16)hp[16 + j];   // K: base+16..+23

#pragma unroll
        for (int nt = 0; nt < 8; ++nt) {   // 8 column tiles of 16
            const _Float16* wp = &WT[(nt * 16 + lrow) * DHID + kt * 32 + hi * 8];
            v8h blo = *(const v8h*)wp;          // 16B LDS load
            v8h bhi = *(const v8h*)(wp + 16);   // K + 16
            v16h b = __builtin_shufflevector(blo, bhi,
                        0,1,2,3,4,5,6,7,8,9,10,11,12,13,14,15);
            acc[nt] = __builtin_amdgcn_wmma_f32_16x16x32_f16(
                          false, a, false, b, (short)0, acc[nt], false, false);
        }
    }

    // ---- store: C/D layout, VGPR r -> row r (lanes 0-15) / r+8 (lanes 16-31)
#pragma unroll
    for (int nt = 0; nt < 8; ++nt) {
#pragma unroll
        for (int r = 0; r < 8; ++r) {
            int orow = tile * 16 + r + hi * 8;
            if (orow < nrows)
                Mout[(size_t)orow * DHID + nt * 16 + lrow] = acc[nt][r];
        }
    }
}

// ----------------------------- edge scatter --------------------------------
// One wave per edge (incl. N virtual self-loop edges appended after E).
// Each lane handles 4 channels: gather msg (L2 hit), scale by norm,
// 4x global_atomic_add_f32 into the accumulator (L2 atomics).

__global__ void __launch_bounds__(256)
k_scatter(const float* __restrict__ Msrc,
          const int* __restrict__ srcIdx, const int* __restrict__ dstIdx,
          const float* __restrict__ dinv, float* __restrict__ Out,
          int e_cnt, int n_cnt) {
    const int e    = blockIdx.x * 8 + (threadIdx.x >> 5);
    const int lane = threadIdx.x & 31;
    const int ea   = e_cnt + n_cnt;
    if (e >= ea) return;

    int s, d;
    if (e < e_cnt) { s = srcIdx[e]; d = dstIdx[e]; }
    else           { s = d = e - e_cnt; }

    const float nrm = dinv[s] * dinv[d];
    const float4 m = *(const float4*)(Msrc + (size_t)s * DHID + lane * 4);
    float* o = Out + (size_t)d * DHID + lane * 4;
    atomicAdd(o + 0, m.x * nrm);
    atomicAdd(o + 1, m.y * nrm);
    atomicAdd(o + 2, m.z * nrm);
    atomicAdd(o + 3, m.w * nrm);
}

// ------------------------------- pooling -----------------------------------

__global__ void k_count(const int* __restrict__ batch, float* __restrict__ counts, int n) {
    int i = blockIdx.x * blockDim.x + threadIdx.x;
    if (i < n) atomicAdd(&counts[batch[i]], 1.0f);
}

__global__ void k_pool(const float* __restrict__ h, const int* __restrict__ batch,
                       float* __restrict__ hg, int n) {
    int i = blockIdx.x * blockDim.x + threadIdx.x;   // over N*128
    if (i < n) {
        int node = i >> 7, c = i & 127;
        atomicAdd(&hg[batch[node] * DHID + c], h[i]);
    }
}

// ------------------------------ head + softmax -----------------------------

__global__ void __launch_bounds__(32)
k_head(const float* __restrict__ hg, const float* __restrict__ counts,
       const float* __restrict__ Wl, const float* __restrict__ bl,
       float* __restrict__ out) {
    const int g = blockIdx.x;
    const int c = threadIdx.x;
    __shared__ float logit[DOUT];

    const float inv = 1.0f / fmaxf(counts[g], 1.0f);
    if (c < DOUT) {
        float acc = bl[c];
        for (int k = 0; k < DHID; ++k)
            acc += hg[g * DHID + k] * inv * Wl[k * DOUT + c];
        logit[c] = acc;
    }
    __syncthreads();
    if (c < DOUT) {
        float mx = -3.0e38f;
        for (int j = 0; j < DOUT; ++j) mx = fmaxf(mx, logit[j]);
        float ssum = 0.0f;
        for (int j = 0; j < DOUT; ++j) ssum += __expf(logit[j] - mx);
        out[g * DOUT + c] = __expf(logit[c] - mx) / ssum;
    }
}

// ------------------------------- launcher ----------------------------------

extern "C" void kernel_launch(void* const* d_in, const int* in_sizes, int n_in,
                              void* d_out, int out_size, void* d_ws, size_t ws_size,
                              hipStream_t stream) {
    const float* x   = (const float*)d_in[0];
    const int*   ei  = (const int*)d_in[1];          // (2, E) row-major
    const int*   bat = (const int*)d_in[2];
    const float* W1  = (const float*)d_in[3];
    const float* b1  = (const float*)d_in[4];
    const float* W2  = (const float*)d_in[5];
    const float* b2  = (const float*)d_in[6];
    const float* W3  = (const float*)d_in[7];
    const float* b3  = (const float*)d_in[8];
    const float* Wl  = (const float*)d_in[9];
    const float* bl  = (const float*)d_in[10];
    float* out = (float*)d_out;

    const int* srcIdx = ei;            // edge_index[0]
    const int* dstIdx = ei + NEDGES;   // edge_index[1]

    // workspace carve-up (ping-pong activations stay L2-resident: ~103 MB)
    float* P      = (float*)d_ws;                       // N*128  (GEMM output / msgs)
    float* Q      = P + (size_t)NNODES * DHID;          // N*128  (scatter accumulator)
    float* dinv   = Q + (size_t)NNODES * DHID;          // N
    float* hg     = dinv + NNODES;                      // G*128
    float* counts = hg + NGRAPH * DHID;                 // G

    const int T = 256;
    const int gN    = (NNODES + T - 1) / T;
    const int gE    = (NEDGES + T - 1) / T;
    const int gNC   = (NNODES * DHID + T - 1) / T;
    const int ntile = (NNODES + 15) / 16;
    const int gGemm = (ntile + 7) / 8;
    const int gScat = (NEDGES + NNODES + 7) / 8;

    // ---- degrees -> dinv (deg seeded at 1.0 for the self loop)
    k_fill<<<gN, T, 0, stream>>>(dinv, 1.0f, NNODES);
    k_degree<<<gE, T, 0, stream>>>(dstIdx, dinv, NEDGES);
    k_rsqrt_inplace<<<gN, T, 0, stream>>>(dinv, NNODES);

    // ---- layer 1: P = x@W1 ; Q = b1 + scatter(P) ; relu
    k_gemm_wmma<<<gGemm, T, 0, stream>>>(x, W1, P, NNODES);
    k_bias_init<<<gNC, T, 0, stream>>>(Q, b1, NNODES * DHID);
    k_scatter<<<gScat, T, 0, stream>>>(P, srcIdx, dstIdx, dinv, Q, NEDGES, NNODES);
    k_relu<<<gNC, T, 0, stream>>>(Q, NNODES * DHID);

    // ---- layer 2
    k_gemm_wmma<<<gGemm, T, 0, stream>>>(Q, W2, P, NNODES);
    k_bias_init<<<gNC, T, 0, stream>>>(Q, b2, NNODES * DHID);
    k_scatter<<<gScat, T, 0, stream>>>(P, srcIdx, dstIdx, dinv, Q, NEDGES, NNODES);
    k_relu<<<gNC, T, 0, stream>>>(Q, NNODES * DHID);

    // ---- layer 3 (no relu)
    k_gemm_wmma<<<gGemm, T, 0, stream>>>(Q, W3, P, NNODES);
    k_bias_init<<<gNC, T, 0, stream>>>(Q, b3, NNODES * DHID);
    k_scatter<<<gScat, T, 0, stream>>>(P, srcIdx, dstIdx, dinv, Q, NEDGES, NNODES);

    // ---- global mean pool + head
    k_fill<<<(NGRAPH * DHID + T - 1) / T, T, 0, stream>>>(hg, 0.0f, NGRAPH * DHID);
    k_fill<<<1, T, 0, stream>>>(counts, 0.0f, NGRAPH);
    k_count<<<gN, T, 0, stream>>>(bat, counts, NNODES);
    k_pool<<<gNC, T, 0, stream>>>(Q, bat, hg, NNODES * DHID);
    k_head<<<NGRAPH, 32, 0, stream>>>(hg, counts, Wl, bl, out);
}
